// PAModule_58737972740597
// MI455X (gfx1250) — matrix operand order repaired and checked by reading, not compiled, not run on hardware
//
#include <hip/hip_runtime.h>

typedef __bf16 bf16;
typedef __attribute__((ext_vector_type(16))) __bf16 v16bf;
typedef __attribute__((ext_vector_type(8)))  __bf16 v8bf;
typedef __attribute__((ext_vector_type(8)))  float  v8f;

#define B_   8
#define C_   256
#define CQ   32
#define N_   4096
#define OTOT 320   // 32 (q) + 32 (k) + 256 (v) output channels
#define NCH  (N_ / 32)

// CDNA5 async global->LDS copies (ASYNCcnt-tracked), if this toolchain exposes them.
#if defined(__AMDGCN__) && __has_builtin(__builtin_amdgcn_global_load_async_to_lds_b128) && \
    __has_builtin(__builtin_amdgcn_s_wait_asynccnt)
#define USE_ASYNC 1
// exact pointer types per hipcc diagnostic: b128 payload = int __vector_size__(16)
typedef int v4i_ __attribute__((vector_size(16)));
typedef __attribute__((address_space(1))) v4i_ as1_v4i;   // global src
typedef __attribute__((address_space(3))) v4i_ as3_v4i;   // LDS dst
#else
#define USE_ASYNC 0
#endif

// ---------- helpers ----------

// A-operand fragment (16x32 bf16): per ISA 7.12.2, lane L (hi = L>>4) holds
// K = {hi*8 .. hi*8+7} in elements 0..7 and K = {16+hi*8 .. 23+hi*8} in 8..15.
__device__ __forceinline__ v16bf load_afrag(const bf16* row, int hi) {
    v8bf a0 = *(const v8bf*)(row + hi * 8);
    v8bf a1 = *(const v8bf*)(row + 16 + hi * 8);
    v16bf r;
#pragma unroll
    for (int e = 0; e < 8; ++e) { r[e] = a0[e]; r[8 + e] = a1[e]; }
    return r;
}

// B-operand fragment (32x16 bf16): lane L holds column n' = L%16, with
// K = {hi*8 .. hi*8+7} in elements 0..7 and K = {16+hi*8 .. 23+hi*8} in 8..15.
__device__ __forceinline__ v16bf load_bfrag(const bf16* row, int hi) {
    v8bf b0 = *(const v8bf*)(row + hi * 8);
    v8bf b1 = *(const v8bf*)(row + 16 + hi * 8);
    v16bf r;
#pragma unroll
    for (int e = 0; e < 8; ++e) { r[e] = b0[e]; r[8 + e] = b1[e]; }
    return r;
}

__device__ __forceinline__ v8f wmma_bf16(v16bf a, v16bf b, v8f c) {
    return __builtin_amdgcn_wmma_f32_16x16x32_bf16(false, a, false, b,
                                                   (short)0, c, false, false);
}

// ---------- kernel 1: pack weights to bf16, biases to f32 ----------
__global__ void prep_weights(const float* __restrict__ wq, const float* __restrict__ bq,
                             const float* __restrict__ wk, const float* __restrict__ bk,
                             const float* __restrict__ wv, const float* __restrict__ bv,
                             bf16* __restrict__ Wb, float* __restrict__ biasf) {
    int idx = blockIdx.x * blockDim.x + threadIdx.x;  // 0 .. 320*256-1
    int o = idx / C_, c = idx % C_;
    float w;
    if (o < 32)       w = wq[o * C_ + c];
    else if (o < 64)  w = wk[(o - 32) * C_ + c];
    else              w = wv[(o - 64) * C_ + c];
    Wb[idx] = (bf16)w;
    if (idx < OTOT) {
        float b = (idx < 32) ? bq[idx] : (idx < 64 ? bk[idx - 32] : bv[idx - 64]);
        biasf[idx] = b;
    }
}

// ---------- kernel 2: x[B][C][N] f32 -> xT[B][N][C] bf16 (tiled transpose) ----------
__global__ void transpose_x(const float* __restrict__ x, bf16* __restrict__ xT) {
    __shared__ float tile[32][33];
    int b  = blockIdx.z;
    int c0 = blockIdx.y * 32;
    int n0 = blockIdx.x * 32;
    int t  = threadIdx.x;
    int nl = t & 31, cl = t >> 5;  // cl in 0..7
    const float* xp = x + ((size_t)b * C_ + c0) * N_ + n0;
#pragma unroll
    for (int s = 0; s < 4; ++s)
        tile[cl + 8 * s][nl] = xp[(size_t)(cl + 8 * s) * N_ + nl];
    __syncthreads();
    bf16* xtp = xT + ((size_t)b * N_ + n0) * C_ + c0;
#pragma unroll
    for (int s = 0; s < 4; ++s)
        xtp[(size_t)(cl + 8 * s) * C_ + nl] = (bf16)tile[nl][cl + 8 * s];
}

// ---------- kernel 3: fused q/k/v 1x1-conv projection via WMMA ----------
// Y[b][n][o] = sum_c xT[b][n][c] * W[o][c] + bias[o]
// o<32 -> q[b][n][o] ; 32<=o<64 -> kT[b][n][o-32] ; o>=64 -> v[b][o-64][n]
__global__ void __launch_bounds__(256) project_qkv(
        const bf16* __restrict__ xT, const bf16* __restrict__ Wb,
        const float* __restrict__ biasf,
        bf16* __restrict__ q, bf16* __restrict__ kT, bf16* __restrict__ v) {
    int lane = threadIdx.x & 31;
    int wave = threadIdx.x >> 5;
    int tid  = blockIdx.x * 8 + wave;               // one 16x16 tile per wave
    int b    = tid / ((N_ / 16) * (OTOT / 16));
    int rem  = tid % ((N_ / 16) * (OTOT / 16));
    int it   = rem / (OTOT / 16);
    int ot   = rem % (OTOT / 16);
    int i0 = it * 16, o0 = ot * 16;
    int lo = lane & 15, hi = lane >> 4;

    const bf16* arow = xT + ((size_t)b * N_ + i0 + lo) * C_;
    const bf16* brow = Wb + (size_t)(o0 + lo) * C_;

    v8f acc = {};
#pragma unroll
    for (int kc = 0; kc < C_ / 32; ++kc) {
        v16bf a = load_afrag(arow + kc * 32, hi);
        v16bf w = load_bfrag(brow + kc * 32, hi);
        acc = wmma_bf16(a, w, acc);
    }
    float bias = biasf[o0 + lo];
#pragma unroll
    for (int r = 0; r < 8; ++r) acc[r] += bias;

    int o = o0 + lo;
    if (o < 64) {
        bf16* dst = (o < 32 ? q : kT) + ((size_t)b * N_ + i0 + 8 * hi) * CQ + (o & 31);
#pragma unroll
        for (int r = 0; r < 8; ++r) dst[(size_t)r * CQ] = (bf16)acc[r];
    } else {
        bf16* dst = v + ((size_t)b * C_ + (o - 64)) * N_ + i0 + 8 * hi;
        v8bf pk;
#pragma unroll
        for (int r = 0; r < 8; ++r) pk[r] = (bf16)acc[r];
        *(v8bf*)dst = pk;
    }
}

// ---------- K/V chunk staging (async-to-LDS when available) ----------
// 32 keys per chunk: kT chunk = 32 rows x 64B (128 x 16B segs, 1 per thread),
// v chunk = 256 rows x 64B (1024 x 16B segs, 8 per thread).
__device__ __forceinline__ void stage_kv(const bf16* kbase, const bf16* vbase, int j0,
                                         int t, bf16 (*kb)[32], bf16 (*vb)[32]) {
    int row = t >> 2, part = t & 3;
#if USE_ASYNC
    __builtin_amdgcn_global_load_async_to_lds_b128(
        (as1_v4i*)(kbase + (size_t)(j0 + row) * CQ + part * 8),
        (as3_v4i*)&kb[row][part * 8], 0, 0);
#pragma unroll
    for (int s = 0; s < 8; ++s) {
        int idx = t + 128 * s;
        int r2 = idx >> 2, p2 = idx & 3;
        __builtin_amdgcn_global_load_async_to_lds_b128(
            (as1_v4i*)(vbase + (size_t)r2 * N_ + j0 + p2 * 8),
            (as3_v4i*)&vb[r2][p2 * 8], 0, 0);
    }
#else
    *(v8bf*)&kb[row][part * 8] = *(const v8bf*)(kbase + (size_t)(j0 + row) * CQ + part * 8);
#pragma unroll
    for (int s = 0; s < 8; ++s) {
        int idx = t + 128 * s;
        int r2 = idx >> 2, p2 = idx & 3;
        *(v8bf*)&vb[r2][p2 * 8] = *(const v8bf*)(vbase + (size_t)r2 * N_ + j0 + p2 * 8);
    }
#endif
}

// ---------- kernel 4: fused flash-attention + residual ----------
// grid: (N/64, B), block 128 (4 waves). Wave w owns 16 query rows i0..i0+15.
// Double-buffered K/V chunks of 32 keys staged in LDS, shared by all 4 waves.
__global__ void __launch_bounds__(128) attention(
        const bf16* __restrict__ q, const bf16* __restrict__ kT,
        const bf16* __restrict__ v, const float* __restrict__ x,
        float* __restrict__ out) {
    __shared__ __align__(128) bf16 kbuf[2][32][32];    // [buf][j_local][d]
    __shared__ __align__(128) bf16 vbuf[2][256][32];   // [buf][c][j_local]
    __shared__ __align__(128) bf16 pbuf[4][16][32];    // per-wave P tile [i][j]

    int t = threadIdx.x;
    int wave = t >> 5, lane = t & 31;
    int lo = lane & 15, hi = lane >> 4;
    int b  = blockIdx.y;
    int i0 = blockIdx.x * 64 + wave * 16;

    // q A-fragment for this wave's 16 rows (K = Cq = 32, loaded once)
    const bf16* qrow = q + ((size_t)b * N_ + i0 + lo) * CQ;
    v16bf qa = load_afrag(qrow, hi);

    v8f accO[16];
#pragma unroll
    for (int ct = 0; ct < 16; ++ct) accO[ct] = (v8f){};
    float mrow[8], lrow[8];
#pragma unroll
    for (int r = 0; r < 8; ++r) { mrow[r] = -1.0e30f; lrow[r] = 0.0f; }

    const bf16* kbase = kT + (size_t)b * N_ * CQ;
    const bf16* vbase = v  + (size_t)b * C_ * N_;

    stage_kv(kbase, vbase, 0, t, kbuf[0], vbuf[0]);   // prefetch chunk 0

    for (int jc = 0; jc < NCH; ++jc) {
        int cur = jc & 1;
        if (jc + 1 < NCH) {
            // issue prefetch of next chunk into the other buffer (freed by the
            // trailing barrier of the previous iteration)
            stage_kv(kbase, vbase, (jc + 1) * 32, t, kbuf[cur ^ 1], vbuf[cur ^ 1]);
#if USE_ASYNC
            // in-order completion: <=9 outstanding => current chunk's 9 landed
            __builtin_amdgcn_s_wait_asynccnt(9);
#endif
        } else {
#if USE_ASYNC
            __builtin_amdgcn_s_wait_asynccnt(0);
#endif
        }
        __syncthreads();  // all waves' copies for chunk jc visible

        bf16 (*kc)[32] = kbuf[cur];
        bf16 (*vc)[32] = vbuf[cur];

        // S = q . k  (two 16x16 tiles covering 32 keys)
        v16bf bk0 = load_bfrag(&kc[lo][0], hi);       // j_local = lo
        v16bf bk1 = load_bfrag(&kc[16 + lo][0], hi);  // j_local = 16+lo
        v8f S0 = wmma_bf16(qa, bk0, (v8f){});
        v8f S1 = wmma_bf16(qa, bk1, (v8f){});

        // online softmax; C/D layout: element (i = r+8*hi, j = lo [+16])
        float alpha[8];
#pragma unroll
        for (int r = 0; r < 8; ++r) {
            float mx = fmaxf(S0[r], S1[r]);
#pragma unroll
            for (int off = 8; off >= 1; off >>= 1)
                mx = fmaxf(mx, __shfl_xor(mx, off, 32)); // stays within 16-lane half
            float nm = fmaxf(mrow[r], mx);
            float al = __expf(mrow[r] - nm);
            float p0 = __expf(S0[r] - nm);
            float p1 = __expf(S1[r] - nm);
            float rs = p0 + p1;
#pragma unroll
            for (int off = 8; off >= 1; off >>= 1)
                rs += __shfl_xor(rs, off, 32);
            mrow[r] = nm;
            lrow[r] = lrow[r] * al + rs;
            alpha[r] = al;
            pbuf[wave][r + 8 * hi][lo]      = (bf16)p0;
            pbuf[wave][r + 8 * hi][16 + lo] = (bf16)p1;
        }
#pragma unroll
        for (int ct = 0; ct < 16; ++ct)
#pragma unroll
            for (int r = 0; r < 8; ++r) accO[ct][r] *= alpha[r];

        // P tile is wave-private; LDS ops from one wave execute in order, so a
        // compiler reordering fence suffices (no cross-wave barrier needed).
        asm volatile("" ::: "memory");

        v16bf pa = load_afrag(&pbuf[wave][lo][0], hi);
#pragma unroll
        for (int ct = 0; ct < 16; ++ct) {
            v16bf bv = load_bfrag(&vc[ct * 16 + lo][0], hi);
            accO[ct] = wmma_bf16(pa, bv, accO[ct]);
        }
        __syncthreads();  // all waves done reading chunk jc buffers
    }

    // epilogue: normalize, add residual, write out[b][c][i]
    float rl[8];
#pragma unroll
    for (int r = 0; r < 8; ++r) rl[r] = 1.0f / lrow[r];
#pragma unroll
    for (int ct = 0; ct < 16; ++ct) {
        int c = ct * 16 + lo;
        const float* xp = x   + ((size_t)b * C_ + c) * N_ + i0 + 8 * hi;
        float*       op = out + ((size_t)b * C_ + c) * N_ + i0 + 8 * hi;
#pragma unroll
        for (int r = 0; r < 8; ++r) op[r] = xp[r] + accO[ct][r] * rl[r];
    }
}

// ---------- launcher ----------
extern "C" void kernel_launch(void* const* d_in, const int* in_sizes, int n_in,
                              void* d_out, int out_size, void* d_ws, size_t ws_size,
                              hipStream_t stream) {
    const float* x  = (const float*)d_in[0];
    const float* wq = (const float*)d_in[1];
    const float* bq = (const float*)d_in[2];
    const float* wk = (const float*)d_in[3];
    const float* bk = (const float*)d_in[4];
    const float* wv = (const float*)d_in[5];
    const float* bv = (const float*)d_in[6];
    float* out = (float*)d_out;

    char* ws = (char*)d_ws;
    const size_t OFF_W    = 0;                               // 320*256*2 = 163840
    const size_t OFF_BIAS = 163840;                          // 1280
    const size_t OFF_XT   = 165120;                          // 16 MB
    const size_t OFF_Q    = OFF_XT + (size_t)B_ * N_ * C_ * 2;
    const size_t OFF_K    = OFF_Q  + (size_t)B_ * N_ * CQ * 2;
    const size_t OFF_V    = OFF_K  + (size_t)B_ * N_ * CQ * 2;

    bf16*  Wb    = (bf16*)(ws + OFF_W);
    float* biasf = (float*)(ws + OFF_BIAS);
    bf16*  xT    = (bf16*)(ws + OFF_XT);
    bf16*  qb    = (bf16*)(ws + OFF_Q);
    bf16*  kb    = (bf16*)(ws + OFF_K);
    bf16*  vb    = (bf16*)(ws + OFF_V);

    prep_weights<<<dim3(OTOT), dim3(C_), 0, stream>>>(wq, bq, wk, bk, wv, bv, Wb, biasf);
    transpose_x<<<dim3(N_ / 32, C_ / 32, B_), dim3(256), 0, stream>>>(x, xT);
    project_qkv<<<dim3(B_ * (N_ / 16) * (OTOT / 16) / 8), dim3(256), 0, stream>>>(
        xT, Wb, biasf, qb, kb, vb);
    attention<<<dim3(N_ / 64, B_), dim3(128), 0, stream>>>(qb, kb, vb, x, out);
}
